// PointTransformerEncoder_17214228922883
// MI455X (gfx1250) — matrix operand order repaired and checked by reading, not compiled, not run on hardware
//
#include <hip/hip_runtime.h>
#include <hip/hip_bf16.h>
#include <math.h>

// ---------------------------------------------------------------------------
// Point Transformer encoder for MI455X (gfx1250, wave32, WMMA).
// B=4 S=8 N=1024 F=6 H=64 L=3 K=16 D=3
// Round 2: f16 pre-swizzled weights + f16 activation mirrors -> contiguous
// b128 fragment loads, no f32->f16 conversions in the WMMA hot loops.
// ---------------------------------------------------------------------------

typedef __attribute__((ext_vector_type(16))) _Float16 v16h;
typedef __attribute__((ext_vector_type(8)))  _Float16 v8h;
typedef __attribute__((ext_vector_type(8)))  float    v8f;

#define NTOT   32768        // B*S*N
#define HDIM   64
#define KNN    16
#define NSCENE 32           // B*S
#define NPTS   1024         // N
#define FDIM   6
#define EPSLN  1e-6f

__device__ __forceinline__ float gelu_tanh(float x) {
  const float c0 = 0.7978845608028654f;   // sqrt(2/pi)
  float x3 = x * x * x;
  return 0.5f * x * (1.0f + tanhf(c0 * (x + 0.044715f * x3)));
}

// ---- WMMA fragment helpers (16x16x32 f16 -> f32) ---------------------------
// A (16x32, MxK) from a row-major f16 matrix: lane<16 -> row=lane,
//   elems 0..7 = K ko..ko+7, elems 8..15 = K ko+16..ko+23, ko = k0 (+8 hi half)
__device__ __forceinline__ v16h load_a_frag16(const _Float16* __restrict__ Ah,
                                              int row_base, int k0, int lane) {
  int r  = row_base + (lane & 15);
  int ko = k0 + ((lane >> 4) << 3);
  const _Float16* p = Ah + r * HDIM + ko;
  v8h lo = *(const v8h*)p;          // 16B aligned
  v8h hi = *(const v8h*)(p + 16);   // 16B aligned
  return __builtin_shufflevector(lo, hi, 0, 1, 2, 3, 4, 5, 6, 7,
                                 8, 9, 10, 11, 12, 13, 14, 15);
}

// B fragment from pre-swizzled weight: frag = ks*4 + n_tile, lane-contiguous.
__device__ __forceinline__ v16h load_b_frag16(const _Float16* __restrict__ Wh,
                                              int frag, int lane) {
  return *(const v16h*)(Wh + ((size_t)(frag * 32 + lane) << 4));  // 32B chunk
}

__device__ __forceinline__ v8f wmma16(v16h a, v16h b, v8f c) {
  return __builtin_amdgcn_wmma_f32_16x16x32_f16(false, a, false, b,
                                                (short)0, c, false, false);
}

// ---------------------------------------------------------------------------
// 0) Swizzle one 64x64 f32 weight into f16 WMMA-B fragment order (once).
//    element (k, n): frag = (k>>5)*4 + (n>>4); lane = (n&15) + 16*((k>>3)&1);
//    elem = (k & 7) + 8*((k>>4)&1).
// ---------------------------------------------------------------------------
__global__ void __launch_bounds__(256) convert_w_kernel(
    const float* __restrict__ W, _Float16* __restrict__ Wh) {
  int i = blockIdx.x * 256 + threadIdx.x;        // 4096 elements
  if (i >= HDIM * HDIM) return;
  int k = i >> 6, n = i & 63;
  int frag = ((k >> 5) << 2) + (n >> 4);
  int lane = (n & 15) + (((k >> 3) & 1) << 4);
  int elem = (k & 7) + (((k >> 4) & 1) << 3);
  Wh[((frag * 32 + lane) << 4) + elem] = (_Float16)W[i];
}

// ---------------------------------------------------------------------------
// 1) x = points @ W_in + b_in  (K=6, trivial -> VALU). Writes f32 + f16 mirror.
// ---------------------------------------------------------------------------
__global__ void __launch_bounds__(256) input_proj_kernel(
    const float* __restrict__ points, const float* __restrict__ w_in,
    const float* __restrict__ b_in, float* __restrict__ x,
    _Float16* __restrict__ xh) {
  int i = blockIdx.x * 256 + threadIdx.x;
  if (i >= NTOT * HDIM) return;
  int row = i >> 6, h = i & 63;
  const float* p = points + row * FDIM;
  float acc = b_in[h];
#pragma unroll
  for (int f = 0; f < FDIM; ++f) acc += p[f] * w_in[f * HDIM + h];
  x[i]  = acc;
  xh[i] = (_Float16)acc;
}

// ---------------------------------------------------------------------------
// 2) KNN (computed ONCE: pos is layer-invariant). One block per scene,
//    positions staged in LDS, register insertion-sort top-16.
// ---------------------------------------------------------------------------
__global__ void __launch_bounds__(256) knn_kernel(
    const float* __restrict__ points, int* __restrict__ nidx) {
  __shared__ float sx[NPTS], sy[NPTS], sz[NPTS];
  int scene = blockIdx.x;
  const float* base = points + (size_t)scene * NPTS * FDIM;
  for (int i = threadIdx.x; i < NPTS; i += 256) {
    sx[i] = base[i * FDIM + 0];
    sy[i] = base[i * FDIM + 1];
    sz[i] = base[i * FDIM + 2];
  }
  __syncthreads();
  for (int qo = 0; qo < 4; ++qo) {
    int n = qo * 256 + threadIdx.x;
    float qx = sx[n], qy = sy[n], qz = sz[n];
    float bd[KNN];
    int   bi[KNN];
#pragma unroll
    for (int t = 0; t < KNN; ++t) { bd[t] = 3.4e38f; bi[t] = 0; }
    for (int j = 0; j < NPTS; ++j) {
      float dx = sx[j] - qx, dy = sy[j] - qy, dz = sz[j] - qz;
      float d = dx * dx + dy * dy + dz * dz;
      if (d < bd[KNN - 1]) {               // strict '<' matches jax tie-break
        bd[KNN - 1] = d; bi[KNN - 1] = j;
#pragma unroll
        for (int t = KNN - 1; t > 0; --t) {
          if (bd[t] < bd[t - 1]) {
            float td = bd[t]; bd[t] = bd[t - 1]; bd[t - 1] = td;
            int   ti = bi[t]; bi[t] = bi[t - 1]; bi[t - 1] = ti;
          }
        }
      }
    }
    int* dst = nidx + ((size_t)scene * NPTS + n) * KNN;
#pragma unroll
    for (int t = 0; t < KNN; ++t) dst[t] = scene * NPTS + bi[t];  // global idx
  }
}

// ---------------------------------------------------------------------------
// 3) Fused q/k/v projection + positional encoding (WMMA).
//    Stores qp = q+pe, kp = k+pe, v. Wave = 16x64 output tile, 8 waves/block.
// ---------------------------------------------------------------------------
__global__ void __launch_bounds__(256) qkv_pe_kernel(
    const _Float16* __restrict__ xh, const float* __restrict__ points,
    const _Float16* __restrict__ Wqh, const float* __restrict__ bq,
    const _Float16* __restrict__ Wkh, const float* __restrict__ bk,
    const _Float16* __restrict__ Wvh, const float* __restrict__ bv,
    const float* __restrict__ Wpe, const float* __restrict__ bpe,
    float* __restrict__ qp, float* __restrict__ kp, float* __restrict__ vv) {
  int lane = threadIdx.x & 31;
  int wave = threadIdx.x >> 5;
  int row_base = blockIdx.x * 128 + wave * 16;

  // pull the swizzled weights toward the WGP early (global_prefetch_b8)
  __builtin_prefetch(Wqh + (lane << 7), 0, 1);
  __builtin_prefetch(Wkh + (lane << 7), 0, 1);
  __builtin_prefetch(Wvh + (lane << 7), 0, 1);

  v8f zero = {};
  v8f cq[4], ck[4], cv[4];
#pragma unroll
  for (int t = 0; t < 4; ++t) { cq[t] = zero; ck[t] = zero; cv[t] = zero; }

#pragma unroll
  for (int ks = 0; ks < 2; ++ks) {
    v16h a = load_a_frag16(xh, row_base, ks * 32, lane);
#pragma unroll
    for (int t = 0; t < 4; ++t) {
      int frag = ks * 4 + t;
      cq[t] = wmma16(a, load_b_frag16(Wqh, frag, lane), cq[t]);
      ck[t] = wmma16(a, load_b_frag16(Wkh, frag, lane), ck[t]);
      cv[t] = wmma16(a, load_b_frag16(Wvh, frag, lane), cv[t]);
    }
  }

  // epilogue: bias + pos-enc fused; C layout: reg r -> row r (+8 for hi half)
#pragma unroll
  for (int t = 0; t < 4; ++t) {
    int col = t * 16 + (lane & 15);
    float bqc = bq[col], bkc = bk[col], bvc = bv[col], bpec = bpe[col];
    float w0 = Wpe[col], w1 = Wpe[HDIM + col], w2 = Wpe[2 * HDIM + col];
#pragma unroll
    for (int r = 0; r < 8; ++r) {
      int row = row_base + r + ((lane >> 4) << 3);
      const float* pp = points + row * FDIM;
      float pe = pp[0] * w0 + pp[1] * w1 + pp[2] * w2 + bpec;
      qp[row * HDIM + col] = cq[t][r] + bqc + pe;
      kp[row * HDIM + col] = ck[t][r] + bkc + pe;
      vv[row * HDIM + col] = cv[t][r] + bvc;
    }
  }
}

// ---------------------------------------------------------------------------
// 4) Local attention: wave32 per point, lane owns channels {lane, lane+32}.
//    s_j = sum_h qp*(kp_n + pd_enc)/8 * Wa + ba  via shfl_xor butterfly.
//    Output written directly as f16 (it is only consumed as a WMMA A operand).
// ---------------------------------------------------------------------------
__global__ void __launch_bounds__(256) knn_attn_kernel(
    const float* __restrict__ qp, const float* __restrict__ kp,
    const float* __restrict__ vv, const float* __restrict__ points,
    const int* __restrict__ nidx,
    const float* __restrict__ Wpd, const float* __restrict__ bpd,
    const float* __restrict__ Wa, const float* __restrict__ ba,
    _Float16* __restrict__ attnh) {
  int lane  = threadIdx.x & 31;
  int point = blockIdx.x * 8 + (threadIdx.x >> 5);
  int ch0 = lane, ch1 = lane + 32;
  const float inv_scale = 0.125f;   // 1/sqrt(64)

  float q0 = qp[point * HDIM + ch0];
  float q1 = qp[point * HDIM + ch1];
  const float* pp = points + point * FDIM;
  float px = pp[0], py = pp[1], pz = pp[2];

  float w00 = Wpd[ch0], w10 = Wpd[HDIM + ch0], w20 = Wpd[2 * HDIM + ch0];
  float w01 = Wpd[ch1], w11 = Wpd[HDIM + ch1], w21 = Wpd[2 * HDIM + ch1];
  float b0 = bpd[ch0], b1 = bpd[ch1];
  float wa0 = Wa[ch0], wa1 = Wa[ch1];
  float bav = ba[0];

  float s[KNN], v0[KNN], v1[KNN];
  const int* nb = nidx + (size_t)point * KNN;
#pragma unroll
  for (int j = 0; j < KNN; ++j) {
    int idx = nb[j];
    const float* np = points + idx * FDIM;
    float dx = np[0] - px, dy = np[1] - py, dz = np[2] - pz;
    float pd0 = dx * w00 + dy * w10 + dz * w20 + b0;
    float pd1 = dx * w01 + dy * w11 + dz * w21 + b1;
    float t0 = q0 * (kp[idx * HDIM + ch0] + pd0) * inv_scale;
    float t1 = q1 * (kp[idx * HDIM + ch1] + pd1) * inv_scale;
    v0[j] = vv[idx * HDIM + ch0];
    v1[j] = vv[idx * HDIM + ch1];
    float part = t0 * wa0 + t1 * wa1;
#pragma unroll
    for (int m = 16; m >= 1; m >>= 1) part += __shfl_xor(part, m, 32);
    s[j] = part + bav;
  }
  // softmax over K=16 (replicated in every lane)
  float mx = s[0];
#pragma unroll
  for (int j = 1; j < KNN; ++j) mx = fmaxf(mx, s[j]);
  float sum = 0.0f;
#pragma unroll
  for (int j = 0; j < KNN; ++j) { s[j] = __expf(s[j] - mx); sum += s[j]; }
  float inv = 1.0f / sum;
  float o0 = 0.0f, o1 = 0.0f;
#pragma unroll
  for (int j = 0; j < KNN; ++j) {
    float w = s[j] * inv;
    o0 += w * v0[j];
    o1 += w * v1[j];
  }
  attnh[point * HDIM + ch0] = (_Float16)o0;
  attnh[point * HDIM + ch1] = (_Float16)o1;
}

// ---------------------------------------------------------------------------
// 5) Output projection (WMMA) + GELU + residual + LayerNorm, fused via LDS.
//    In-place update of x (block touches only its own 128 rows); also
//    refreshes the f16 mirror consumed by next layer's q/k/v GEMM.
// ---------------------------------------------------------------------------
__global__ void __launch_bounds__(256) outproj_ln_kernel(
    const _Float16* __restrict__ attnh, float* __restrict__ x,
    _Float16* __restrict__ xh,
    const _Float16* __restrict__ Woh, const float* __restrict__ bo,
    const float* __restrict__ ln_s, const float* __restrict__ ln_b) {
  __shared__ float tile[128][HDIM + 1];
  int lane = threadIdx.x & 31;
  int wave = threadIdx.x >> 5;
  int row_base = blockIdx.x * 128 + wave * 16;

  __builtin_prefetch(Woh + (lane << 7), 0, 1);

  v8f zero = {};
  v8f c[4];
#pragma unroll
  for (int t = 0; t < 4; ++t) c[t] = zero;
#pragma unroll
  for (int ks = 0; ks < 2; ++ks) {
    v16h a = load_a_frag16(attnh, row_base, ks * 32, lane);
#pragma unroll
    for (int t = 0; t < 4; ++t) {
      c[t] = wmma16(a, load_b_frag16(Woh, ks * 4 + t, lane), c[t]);
    }
  }
#pragma unroll
  for (int t = 0; t < 4; ++t) {
    int col = t * 16 + (lane & 15);
    float boc = bo[col];
#pragma unroll
    for (int r = 0; r < 8; ++r) {
      int rl  = wave * 16 + r + ((lane >> 4) << 3);
      int row = blockIdx.x * 128 + rl;
      float h = gelu_tanh(c[t][r] + boc);
      tile[rl][col] = x[row * HDIM + col] + h;   // residual
    }
  }
  __syncthreads();
  // LayerNorm over each of the 128 rows (one thread per row)
  for (int rl = threadIdx.x; rl < 128; rl += 256) {
    float sum = 0.0f, sq = 0.0f;
#pragma unroll
    for (int h = 0; h < HDIM; ++h) { float v = tile[rl][h]; sum += v; sq += v * v; }
    float mu = sum * (1.0f / HDIM);
    float var = sq * (1.0f / HDIM) - mu * mu;
    float rs = rsqrtf(var + EPSLN);
    int row = blockIdx.x * 128 + rl;
#pragma unroll
    for (int h = 0; h < HDIM; ++h) {
      float v = (tile[rl][h] - mu) * rs * ln_s[h] + ln_b[h];
      x[row * HDIM + h]  = v;
      xh[row * HDIM + h] = (_Float16)v;
    }
  }
}

// ---------------------------------------------------------------------------
// 6) Max-pool over N per scene
// ---------------------------------------------------------------------------
__global__ void __launch_bounds__(256) maxpool_kernel(
    const float* __restrict__ x, float* __restrict__ pooled) {
  __shared__ float part[4][HDIM];
  int h = threadIdx.x & 63;
  int c = threadIdx.x >> 6;
  size_t base = (size_t)blockIdx.x * NPTS;
  float m = -3.4e38f;
  for (int n = c * 256; n < (c + 1) * 256; ++n)
    m = fmaxf(m, x[(base + n) * HDIM + h]);
  part[c][h] = m;
  __syncthreads();
  if (threadIdx.x < HDIM) {
    float r = fmaxf(fmaxf(part[0][h], part[1][h]),
                    fmaxf(part[2][h], part[3][h]));
    pooled[blockIdx.x * HDIM + h] = r;
  }
}

// ---------------------------------------------------------------------------
// 7) Tiny S=8 encoder layer + LN + final max-pool. One block per batch.
// ---------------------------------------------------------------------------
__global__ void __launch_bounds__(256) encoder_kernel(
    const float* __restrict__ pooled,
    const float* __restrict__ Wek, const float* __restrict__ bek,
    const float* __restrict__ Weq, const float* __restrict__ beq,
    const float* __restrict__ Wev, const float* __restrict__ bev,
    const float* __restrict__ We1, const float* __restrict__ be1,
    const float* __restrict__ We2, const float* __restrict__ be2,
    const float* __restrict__ ln2_s, const float* __restrict__ ln2_b,
    float* __restrict__ out) {
  __shared__ float xs[8][HDIM], qd[8][HDIM], kd[8][HDIM], vd[8][HDIM];
  __shared__ float t1[8][HDIM], t2[8][HDIM];
  __shared__ float aw[8][8];
  int tid = threadIdx.x, b = blockIdx.x;

  for (int e = tid; e < 512; e += 256) xs[e >> 6][e & 63] = pooled[b * 512 + e];
  __syncthreads();

  for (int e = tid; e < 512; e += 256) {
    int s = e >> 6, h = e & 63;
    float aq = beq[h], ak = bek[h], av = bev[h];
    for (int i = 0; i < HDIM; ++i) {
      float xv = xs[s][i];
      aq += xv * Weq[i * HDIM + h];
      ak += xv * Wek[i * HDIM + h];
      av += xv * Wev[i * HDIM + h];
    }
    qd[s][h] = aq; kd[s][h] = ak; vd[s][h] = av;
  }
  __syncthreads();

  if (tid < 64) {
    int t = tid >> 3, s = tid & 7;
    float a = 0.0f;
    for (int h = 0; h < HDIM; ++h) a += qd[t][h] * kd[s][h];
    aw[t][s] = a * 0.125f;
  }
  __syncthreads();
  if (tid < 8) {
    float mx = aw[tid][0];
    for (int s = 1; s < 8; ++s) mx = fmaxf(mx, aw[tid][s]);
    float sum = 0.0f;
    for (int s = 0; s < 8; ++s) { aw[tid][s] = __expf(aw[tid][s] - mx); sum += aw[tid][s]; }
    float inv = 1.0f / sum;
    for (int s = 0; s < 8; ++s) aw[tid][s] *= inv;
  }
  __syncthreads();

  for (int e = tid; e < 512; e += 256) {
    int t = e >> 6, h = e & 63;
    float a = 0.0f;
    for (int s = 0; s < 8; ++s) a += aw[t][s] * vd[s][h];
    t1[t][h] = a;
  }
  __syncthreads();
  for (int e = tid; e < 512; e += 256) {
    int s = e >> 6, h = e & 63;
    float a = be1[h];
    for (int i = 0; i < HDIM; ++i) a += t1[s][i] * We1[i * HDIM + h];
    t2[s][h] = gelu_tanh(a);
  }
  __syncthreads();
  for (int e = tid; e < 512; e += 256) {
    int s = e >> 6, h = e & 63;
    float a = be2[h];
    for (int i = 0; i < HDIM; ++i) a += t2[s][i] * We2[i * HDIM + h];
    xs[s][h] += a;                 // residual
  }
  __syncthreads();
  if (tid < 8) {
    float sum = 0.0f, sq = 0.0f;
    for (int h = 0; h < HDIM; ++h) { float v = xs[tid][h]; sum += v; sq += v * v; }
    float mu = sum * (1.0f / HDIM);
    float var = sq * (1.0f / HDIM) - mu * mu;
    float rs = rsqrtf(var + EPSLN);
    for (int h = 0; h < HDIM; ++h)
      t1[tid][h] = (xs[tid][h] - mu) * rs * ln2_s[h] + ln2_b[h];
  }
  __syncthreads();
  if (tid < HDIM) {
    float m = t1[0][tid];
    for (int s = 1; s < 8; ++s) m = fmaxf(m, t1[s][tid]);
    out[b * HDIM + tid] = m;
  }
}

// ---------------------------------------------------------------------------
extern "C" void kernel_launch(void* const* d_in, const int* in_sizes, int n_in,
                              void* d_out, int out_size, void* d_ws, size_t ws_size,
                              hipStream_t stream) {
  const float* points = (const float*)d_in[0];
  const float* w_in = (const float*)d_in[1];
  const float* b_in = (const float*)d_in[2];
  const float* Wq = (const float*)d_in[3];
  const float* bq = (const float*)d_in[4];
  const float* Wk = (const float*)d_in[5];
  const float* bk = (const float*)d_in[6];
  const float* Wv = (const float*)d_in[7];
  const float* bv = (const float*)d_in[8];
  const float* Wpe = (const float*)d_in[9];
  const float* bpe = (const float*)d_in[10];
  const float* Wpd = (const float*)d_in[11];
  const float* bpd = (const float*)d_in[12];
  const float* Wa = (const float*)d_in[13];
  const float* ba = (const float*)d_in[14];
  const float* Wo = (const float*)d_in[15];
  const float* bo = (const float*)d_in[16];
  const float* ln_s = (const float*)d_in[17];
  const float* ln_b = (const float*)d_in[18];
  const float* Wek = (const float*)d_in[19];
  const float* bek = (const float*)d_in[20];
  const float* Weq = (const float*)d_in[21];
  const float* beq = (const float*)d_in[22];
  const float* Wev = (const float*)d_in[23];
  const float* bev = (const float*)d_in[24];
  const float* We1 = (const float*)d_in[25];
  const float* be1 = (const float*)d_in[26];
  const float* We2 = (const float*)d_in[27];
  const float* be2 = (const float*)d_in[28];
  const float* ln2_s = (const float*)d_in[29];
  const float* ln2_b = (const float*)d_in[30];

  // ---- workspace layout -----------------------------------------------
  const size_t NH  = (size_t)NTOT * HDIM;       // 2M elements
  const int    WHH = HDIM * HDIM;               // 4096
  char* wsp = (char*)d_ws;
  float* x       = (float*)wsp;                 wsp += NH * sizeof(float);
  float* qp      = (float*)wsp;                 wsp += NH * sizeof(float);
  float* kp      = (float*)wsp;                 wsp += NH * sizeof(float);
  float* vv      = (float*)wsp;                 wsp += NH * sizeof(float);
  float* pooled  = (float*)wsp;                 wsp += NSCENE * HDIM * sizeof(float);
  int*   nidx    = (int*)wsp;                   wsp += (size_t)NTOT * KNN * sizeof(int);
  _Float16* xh    = (_Float16*)wsp;             wsp += NH * sizeof(_Float16);
  _Float16* attnh = (_Float16*)wsp;             wsp += NH * sizeof(_Float16);
  _Float16* whh   = (_Float16*)wsp;             // 12 * 4096 f16 = 96KB

  // ---- one-time weight swizzle (f32 -> f16 fragment order) ------------
  for (int l = 0; l < 3; ++l) {
    convert_w_kernel<<<16, 256, 0, stream>>>(Wq + l * WHH, whh + (l * 4 + 0) * WHH);
    convert_w_kernel<<<16, 256, 0, stream>>>(Wk + l * WHH, whh + (l * 4 + 1) * WHH);
    convert_w_kernel<<<16, 256, 0, stream>>>(Wv + l * WHH, whh + (l * 4 + 2) * WHH);
    convert_w_kernel<<<16, 256, 0, stream>>>(Wo + l * WHH, whh + (l * 4 + 3) * WHH);
  }

  input_proj_kernel<<<(NTOT * HDIM + 255) / 256, 256, 0, stream>>>(
      points, w_in, b_in, x, xh);
  knn_kernel<<<NSCENE, 256, 0, stream>>>(points, nidx);  // once: pos invariant

  for (int l = 0; l < 3; ++l) {
    qkv_pe_kernel<<<NTOT / 128, 256, 0, stream>>>(
        xh, points,
        whh + (l * 4 + 0) * WHH, bq + l * HDIM,
        whh + (l * 4 + 1) * WHH, bk + l * HDIM,
        whh + (l * 4 + 2) * WHH, bv + l * HDIM,
        Wpe + l * 3 * HDIM, bpe + l * HDIM,
        qp, kp, vv);
    knn_attn_kernel<<<NTOT / 8, 256, 0, stream>>>(
        qp, kp, vv, points, nidx,
        Wpd + l * 3 * HDIM, bpd + l * HDIM, Wa + l * HDIM, ba + l,
        attnh);
    outproj_ln_kernel<<<NTOT / 128, 256, 0, stream>>>(
        attnh, x, xh,
        whh + (l * 4 + 3) * WHH, bo + l * HDIM,
        ln_s + l * HDIM, ln_b + l * HDIM);
  }

  maxpool_kernel<<<NSCENE, 256, 0, stream>>>(x, pooled);
  encoder_kernel<<<4, 256, 0, stream>>>(
      pooled, Wek, bek, Weq, beq, Wev, bev, We1, be1, We2, be2,
      ln2_s, ln2_b, (float*)d_out);
}
